// CognitiveRouter_38783554683018
// MI455X (gfx1250) — matrix operand order repaired and checked by reading, not compiled, not run on hardware
//
#include <hip/hip_runtime.h>

#define D_DIM 1536
#define N_LOGITS 20   // 16 experts + 4 modules

typedef _Float16 v16h __attribute__((ext_vector_type(16)));
typedef _Float16 v8h  __attribute__((ext_vector_type(8)));
typedef float    v8f  __attribute__((ext_vector_type(8)));

// ---------------------------------------------------------------------------
// Kernel 1: skinny GEMM  logits[T x 20] = H[T x D] * [We;Wm]^T  via WMMA f16
// One wave per 16-token tile; weights cached in LDS as f16.
// ---------------------------------------------------------------------------
__global__ __launch_bounds__(256) void router_gemm(
    const float* __restrict__ H,    // (T, D)
    const float* __restrict__ Wm,   // (4, D)
    const float* __restrict__ We,   // (16, D)
    float* __restrict__ logits)     // (T, 20): [0..15]=experts, [16..19]=modules
{
    // rows 0..15 = We, rows 16..19 = Wm, converted to f16 once per block
    __shared__ _Float16 Wh[N_LOGITS * D_DIM];

    const int tid = threadIdx.x;
    for (int i = tid; i < N_LOGITS * D_DIM; i += 256) {
        int r = i / D_DIM;
        int c = i - r * D_DIM;
        float v = (r < 16) ? We[r * D_DIM + c] : Wm[(r - 16) * D_DIM + c];
        Wh[i] = (_Float16)v;
    }
    __syncthreads();

    const int lane = tid & 31;
    const int wave = tid >> 5;
    const int n    = lane & 15;   // matrix row (A) / column (B,D)
    const int hi   = lane >> 4;   // half-wave selector

    const long tile      = (long)blockIdx.x * 8 + wave;
    const long tokenBase = tile * 16;

    const float*    __restrict__ hrow = H + (tokenBase + n) * (long)D_DIM;
    const _Float16* __restrict__ bExp = &Wh[n * D_DIM];                 // expert col n
    const _Float16* __restrict__ bMod = &Wh[(16 + (n & 3)) * D_DIM];    // module (dup n>=4)

    v8f accE = {0.f, 0.f, 0.f, 0.f, 0.f, 0.f, 0.f, 0.f};
    v8f accM = {0.f, 0.f, 0.f, 0.f, 0.f, 0.f, 0.f, 0.f};

    for (int k0 = 0; k0 < D_DIM; k0 += 32) {
        // ---- A operand: 16x32 f16. Lane holds K {hi*16 .. hi*16+15} contiguous
        const int ka = k0 + hi * 16;
        float4 a0 = *(const float4*)(hrow + ka + 0);
        float4 a1 = *(const float4*)(hrow + ka + 4);
        float4 a2 = *(const float4*)(hrow + ka + 8);
        float4 a3 = *(const float4*)(hrow + ka + 12);
        v16h A;
        A[0]  = (_Float16)a0.x; A[1]  = (_Float16)a0.y;
        A[2]  = (_Float16)a0.z; A[3]  = (_Float16)a0.w;
        A[4]  = (_Float16)a1.x; A[5]  = (_Float16)a1.y;
        A[6]  = (_Float16)a1.z; A[7]  = (_Float16)a1.w;
        A[8]  = (_Float16)a2.x; A[9]  = (_Float16)a2.y;
        A[10] = (_Float16)a2.z; A[11] = (_Float16)a2.w;
        A[12] = (_Float16)a3.x; A[13] = (_Float16)a3.y;
        A[14] = (_Float16)a3.z; A[15] = (_Float16)a3.w;

        // ---- B operands: 32x16 f16. Lane holds col n, K {hi*16 .. hi*16+15} contiguous
        const int kb = k0 + hi * 16;
        v8h e0 = *(const v8h*)(bExp + kb);
        v8h e1 = *(const v8h*)(bExp + kb + 8);
        v8h m0 = *(const v8h*)(bMod + kb);
        v8h m1 = *(const v8h*)(bMod + kb + 8);
        v16h BE, BM;
#pragma unroll
        for (int i = 0; i < 8; ++i) {
            BE[i] = e0[i]; BE[i + 8] = e1[i];
            BM[i] = m0[i]; BM[i + 8] = m1[i];
        }

        accE = __builtin_amdgcn_wmma_f32_16x16x32_f16(false, A, false, BE,
                                                      (short)0, accE, false, false);
        accM = __builtin_amdgcn_wmma_f32_16x16x32_f16(false, A, false, BM,
                                                      (short)0, accM, false, false);
    }

    // D layout: VGPR v -> token row (v + 8*hi), col n
#pragma unroll
    for (int v = 0; v < 8; ++v) {
        long t = tokenBase + v + 8 * hi;
        logits[t * N_LOGITS + n] = accE[v];
        if (n < 4) logits[t * N_LOGITS + 16 + n] = accM[v];
    }
}

// ---------------------------------------------------------------------------
// Kernel 2: softmaxes + combined probs + top-4 + normalize. 1 thread / token.
// ---------------------------------------------------------------------------
__global__ __launch_bounds__(256) void router_finish(
    const float* __restrict__ logits,  // (T, 20)
    float* __restrict__ outC,          // (T, 16) combined probs
    float* __restrict__ outW,          // (T, 4)  normalized top-k weights
    int*   __restrict__ outI,          // (T, 4)  top-k indices
    int T)
{
    int t = blockIdx.x * blockDim.x + threadIdx.x;
    if (t >= T) return;
    const float* lg = logits + (long)t * N_LOGITS;

    // module softmax over lg[16..19]
    float mmax = lg[16];
#pragma unroll
    for (int i = 1; i < 4; ++i) mmax = fmaxf(mmax, lg[16 + i]);
    float mp[4], msum = 0.f;
#pragma unroll
    for (int i = 0; i < 4; ++i) { mp[i] = __expf(lg[16 + i] - mmax); msum += mp[i]; }
    float minv = 1.0f / msum;

    // per-module expert softmax, fused with module prob
    float comb[16];
#pragma unroll
    for (int g = 0; g < 4; ++g) {
        float gmax = lg[g * 4];
#pragma unroll
        for (int j = 1; j < 4; ++j) gmax = fmaxf(gmax, lg[g * 4 + j]);
        float p[4], s = 0.f;
#pragma unroll
        for (int j = 0; j < 4; ++j) { p[j] = __expf(lg[g * 4 + j] - gmax); s += p[j]; }
        float scale = mp[g] * minv / s;
#pragma unroll
        for (int j = 0; j < 4; ++j) comb[g * 4 + j] = p[j] * scale;
    }
#pragma unroll
    for (int e = 0; e < 16; ++e) outC[(long)t * 16 + e] = comb[e];

    // top-4 (stable: strict '>' keeps lowest index on ties, like jax.lax.top_k)
    bool used[16];
#pragma unroll
    for (int i = 0; i < 16; ++i) used[i] = false;
    int   idx[4];
    float w[4];
    float wsum = 0.f;
#pragma unroll
    for (int j = 0; j < 4; ++j) {
        float best = -1.0f; int bi = 0;
        for (int i = 0; i < 16; ++i)
            if (!used[i] && comb[i] > best) { best = comb[i]; bi = i; }
        used[bi] = true; idx[j] = bi; w[j] = best; wsum += best;
    }
    float ninv = 1.0f / (wsum + 1e-8f);
#pragma unroll
    for (int j = 0; j < 4; ++j) {
        outW[(long)t * 4 + j] = w[j] * ninv;
        outI[(long)t * 4 + j] = idx[j];
    }
}

// ---------------------------------------------------------------------------
extern "C" void kernel_launch(void* const* d_in, const int* in_sizes, int n_in,
                              void* d_out, int out_size, void* d_ws, size_t ws_size,
                              hipStream_t stream)
{
    const float* H  = (const float*)d_in[0];   // (T, 1536)
    const float* Wm = (const float*)d_in[1];   // (4, 1536)
    const float* We = (const float*)d_in[2];   // (16, 1536)
    const int T = in_sizes[0] / D_DIM;

    float* logits = (float*)d_ws;              // (T, 20) scratch
    float* outC = (float*)d_out;               // (T, 16)
    float* outW = outC + (long)T * 16;         // (T, 4)
    int*   outI = (int*)(outW + (long)T * 4);  // (T, 4)

    const int tiles  = T / 16;                 // 16 tokens per wave
    const int blocks = (tiles + 7) / 8;        // 8 waves per block

    router_gemm<<<blocks, 256, 0, stream>>>(H, Wm, We, logits);
    router_finish<<<(T + 255) / 256, 256, 0, stream>>>(logits, outC, outW, outI, T);
}
